// GRUEnc_2791728742990
// MI455X (gfx1250) — matrix operand order repaired and checked by reading, compiled, not run on hardware
//
#include <hip/hip_runtime.h>
#include <hip/hip_bf16.h>

// ---------------------------------------------------------------------------
// GRU encoder for MI455X (gfx1250, wave32, WMMA).
//   B=32768, IN=256, H=512, seq_len=8 (fixed by setup_inputs).
//
// Roofline: 8 sequential (32768x512)@(512x1536) GEMMs dominate (412 GFLOP).
// Working set (h 32MB f16 + gi 96MB f16 + weights 1.5MB) fits the 192MB L2,
// so the recurrence is matrix-math / L2-bound -> v_wmma_f32_16x16x32_f16
// with f32 accumulation, and 32x32 register tiles per wave to maximize
// fragment reuse (12 WMMA : 16 b128-loads per K-step; A reused x6, B x2).
//
// Workspace (~180 MB): X_f16 16MB | W*_f16 2.6MB | gi_f16 96MB |
//                      h_f16 ping-pong 64MB | bit[8] 1MB
// ---------------------------------------------------------------------------

typedef __attribute__((ext_vector_type(16))) _Float16 v16h;
typedef __attribute__((ext_vector_type(8)))  _Float16 v8h;
typedef __attribute__((ext_vector_type(8)))  float    v8f;

#define BATCH   32768
#define INDIM   256
#define HDIM    512
#define GDIM    1536   /* 3*HDIM */
#define SEQ     8

__device__ __forceinline__ v8h ld8(const _Float16* p) {
  return *(const v8h*)p;   // 16B aligned by construction
}
__device__ __forceinline__ v16h concat8(v8h a, v8h b) {
  return __builtin_shufflevector(a, b, 0,1,2,3,4,5,6,7,8,9,10,11,12,13,14,15);
}
__device__ __forceinline__ float sigmoidf_(float x) {
  return 1.0f / (1.0f + __expf(-x));
}

// ---------------------------------------------------------------------------
// Prep: fp32 -> fp16 conversions + w_bit extraction (W_ih column 0).
// ---------------------------------------------------------------------------
__global__ __launch_bounds__(256)
void prep_kernel(const float* __restrict__ Wproj_f,  // 512 x 256
                 const float* __restrict__ Wih_f,    // 1536 x 257
                 const float* __restrict__ Whh_f,    // 1536 x 512
                 const float* __restrict__ X_f,      // B x 256
                 _Float16* __restrict__ Wproj_h,
                 _Float16* __restrict__ Wih_h,       // 1536 x 256 (cols 1..256)
                 _Float16* __restrict__ Whh_h,
                 float*    __restrict__ wbit,        // 1536
                 _Float16* __restrict__ X_h)
{
  const long NP = 512L * 256, NI = 1536L * 256, NH = 1536L * 512, NB = 1536;
  const long NX = (long)BATCH * 256;
  const long total = NP + NI + NH + NB + NX;
  for (long i = blockIdx.x * (long)blockDim.x + threadIdx.x; i < total;
       i += (long)gridDim.x * blockDim.x) {
    if (i < NP) {
      Wproj_h[i] = (_Float16)Wproj_f[i];
    } else if (i < NP + NI) {
      long j = i - NP; long r = j / 256, c = j % 256;
      Wih_h[j] = (_Float16)Wih_f[r * 257 + 1 + c];
    } else if (i < NP + NI + NH) {
      long j = i - NP - NI;
      Whh_h[j] = (_Float16)Whh_f[j];
    } else if (i < NP + NI + NH + NB) {
      long j = i - NP - NI - NH;
      wbit[j] = Wih_f[j * 257];
    } else {
      long j = i - NP - NI - NH - NB;
      X_h[j] = (_Float16)X_f[j];
    }
  }
}

// ---------------------------------------------------------------------------
// Fused h0 + gi_const GEMM: [B x 256] @ [2048 x 256]^T, K=256, WMMA f16.
// Each wave computes a 32x32 D tile (acc[2][2], 4 WMMA per K-step).
//   n in [0,512)    -> h0 = X@W_proj.T + b_proj
//   n in [512,2048) -> gi = X@W_ih.T   + b_ih
// (a 32-wide N tile never straddles n=512: both are multiples of 32)
// ---------------------------------------------------------------------------
__global__ __launch_bounds__(128)
void proj_gi_kernel(const _Float16* __restrict__ X,
                    const _Float16* __restrict__ Wproj,
                    const _Float16* __restrict__ Wih,
                    const float* __restrict__ b_proj,
                    const float* __restrict__ b_ih,
                    _Float16* __restrict__ h0,
                    _Float16* __restrict__ gi)
{
  const int lane = threadIdx.x & 31;
  const int wave = threadIdx.x >> 5;
  const int hf   = lane >> 4;     // K-half selector (A/B layouts, CDNA5 ISA)
  const int l16  = lane & 15;
  const int m0 = (blockIdx.x * 4 + wave) * 32;
  const int n0 = blockIdx.y * 32;             // 0..2047 (step 32)

  const _Float16* Wbase = (n0 < HDIM) ? Wproj : Wih;
  const int nrel = (n0 < HDIM) ? n0 : (n0 - HDIM);

  const _Float16* arow[2];
  const _Float16* brow[2];
#pragma unroll
  for (int s = 0; s < 2; ++s)
    arow[s] = X + (size_t)(m0 + s * 16 + l16) * INDIM + hf * 16;
#pragma unroll
  for (int nn = 0; nn < 2; ++nn)
    brow[nn] = Wbase + (size_t)(nrel + nn * 16 + l16) * INDIM + hf * 16;

  v8f acc[2][2] = {};
#pragma unroll 2
  for (int k = 0; k < INDIM; k += 32) {
    v16h a[2], b[2];
#pragma unroll
    for (int s = 0; s < 2; ++s)
      a[s] = concat8(ld8(arow[s] + k), ld8(arow[s] + k + 8));
#pragma unroll
    for (int nn = 0; nn < 2; ++nn)
      b[nn] = concat8(ld8(brow[nn] + k), ld8(brow[nn] + k + 8));
#pragma unroll
    for (int s = 0; s < 2; ++s)
#pragma unroll
      for (int nn = 0; nn < 2; ++nn)
        acc[s][nn] = __builtin_amdgcn_wmma_f32_16x16x32_f16(
            false, a[s], false, b[nn], (short)0, acc[s][nn], false, false);
  }

  float bias[2];
#pragma unroll
  for (int nn = 0; nn < 2; ++nn) {
    const int col = n0 + nn * 16 + l16;
    bias[nn] = (n0 < HDIM) ? b_proj[col] : b_ih[col - HDIM];
  }

#pragma unroll
  for (int s = 0; s < 2; ++s) {
#pragma unroll
    for (int v = 0; v < 8; ++v) {
      const int row = m0 + s * 16 + v + hf * 8;  // C/D: lanes16-31 -> M+8
#pragma unroll
      for (int nn = 0; nn < 2; ++nn) {
        const int col = n0 + nn * 16 + l16;
        const float val = acc[s][nn][v] + bias[nn];
        if (n0 < HDIM) h0[(size_t)row * HDIM + col] = (_Float16)val;
        else           gi[(size_t)row * GDIM + (col - HDIM)] = (_Float16)val;
      }
    }
  }
}

// ---------------------------------------------------------------------------
// One GRU step. Each wave owns a 32(M)x32(N) tile of h columns and runs
// 3 gate chains (r/z/n) -> 12 accumulators, 12 WMMA : 16 b128-loads per
// K-step. gh[m,n] = sum_k h[m,k]*W_hh[n+g*512,k] (row-major B direct).
// ---------------------------------------------------------------------------
__global__ __launch_bounds__(128)
void gru_step_kernel(const _Float16* __restrict__ h_in,   // B x 512
                     _Float16* __restrict__ h_out,        // B x 512
                     const _Float16* __restrict__ gi,     // B x 1536
                     const _Float16* __restrict__ Whh,    // 1536 x 512 (f16)
                     const float* __restrict__ wbit,      // 1536
                     const float* __restrict__ b_hh,      // 1536
                     const float* __restrict__ bit_prev,  // B (ignored if !has_prev)
                     int has_prev)
{
  const int lane = threadIdx.x & 31;
  const int wave = threadIdx.x >> 5;
  const int hf   = lane >> 4;
  const int l16  = lane & 15;
  const int m0 = (blockIdx.x * 4 + wave) * 32;
  const int n0 = blockIdx.y * 32;

  const _Float16* arow[2];
#pragma unroll
  for (int s = 0; s < 2; ++s)
    arow[s] = h_in + (size_t)(m0 + s * 16 + l16) * HDIM + hf * 16;

  const _Float16* brow[3][2];
#pragma unroll
  for (int g = 0; g < 3; ++g)
#pragma unroll
    for (int nn = 0; nn < 2; ++nn)
      brow[g][nn] = Whh + (size_t)(g * HDIM + n0 + nn * 16 + l16) * HDIM + hf * 16;

  v8f acc[3][2][2] = {};   // [gate][m-sub][n-sub]

#pragma unroll 2
  for (int k = 0; k < HDIM; k += 32) {
    v16h a[2];
#pragma unroll
    for (int s = 0; s < 2; ++s) {
      if (k + 32 < HDIM) __builtin_prefetch(arow[s] + k + 32, 0, 0);
      a[s] = concat8(ld8(arow[s] + k), ld8(arow[s] + k + 8));
    }
    v16h b[3][2];
#pragma unroll
    for (int g = 0; g < 3; ++g)
#pragma unroll
      for (int nn = 0; nn < 2; ++nn)
        b[g][nn] = concat8(ld8(brow[g][nn] + k), ld8(brow[g][nn] + k + 8));
#pragma unroll
    for (int g = 0; g < 3; ++g)
#pragma unroll
      for (int s = 0; s < 2; ++s)
#pragma unroll
        for (int nn = 0; nn < 2; ++nn)
          acc[g][s][nn] = __builtin_amdgcn_wmma_f32_16x16x32_f16(
              false, a[s], false, b[g][nn], (short)0, acc[g][s][nn], false, false);
  }

  // Per-lane column constants (independent of row) hoisted out of row loop.
  float bh[3][2], wb[3][2], colv[2];
#pragma unroll
  for (int nn = 0; nn < 2; ++nn) {
    const int col = n0 + nn * 16 + l16;
    colv[nn] = (float)col;
#pragma unroll
    for (int g = 0; g < 3; ++g) {
      bh[g][nn] = b_hh[col + g * HDIM];
      wb[g][nn] = wbit[col + g * HDIM];
    }
  }

#pragma unroll
  for (int s = 0; s < 2; ++s) {
#pragma unroll
    for (int v = 0; v < 8; ++v) {
      const int row = m0 + s * 16 + v + hf * 8;
      const float cb = has_prev ? sigmoidf_(bit_prev[row]) : 0.0f;
      const _Float16* girow = gi + (size_t)row * GDIM;
      const _Float16* hrow  = h_in + (size_t)row * HDIM;
#pragma unroll
      for (int nn = 0; nn < 2; ++nn) {
        const int col = n0 + nn * 16 + l16;
        const float ir  = (float)girow[col]            + cb * wb[0][nn];
        const float iz  = (float)girow[col + HDIM]     + cb * wb[1][nn];
        const float in_ = (float)girow[col + 2 * HDIM] + cb * wb[2][nn];
        const float r = sigmoidf_(ir + acc[0][s][nn][v] + bh[0][nn]);
        const float z = sigmoidf_(iz + acc[1][s][nn][v] + bh[1][nn]);
        const float n = tanhf(in_ + r * (acc[2][s][nn][v] + bh[2][nn]));
        const float hold = (float)hrow[col];
        const float hnew = (1.0f - z) * n + z * hold;
        h_out[(size_t)row * HDIM + col] = (_Float16)hnew;
      }
    }
  }
  (void)colv;
}

// ---------------------------------------------------------------------------
// bit_l[row] = h_new[row,:] . W_out + b_out   (wave per row; deterministic,
// no float atomics -> bitwise identical across graph replays).
// ---------------------------------------------------------------------------
__global__ __launch_bounds__(256)
void bit_kernel(const _Float16* __restrict__ h,
                const float* __restrict__ Wout,   // 512
                const float* __restrict__ b_out,  // 1
                float* __restrict__ bit)          // B
{
  const int lane = threadIdx.x & 31;
  const int wave = threadIdx.x >> 5;
  const int row  = blockIdx.x * 8 + wave;
  const _Float16* hr = h + (size_t)row * HDIM;
  float s = 0.0f;
#pragma unroll
  for (int j = lane; j < HDIM; j += 32) s += (float)hr[j] * Wout[j];
#pragma unroll
  for (int o = 16; o; o >>= 1) s += __shfl_xor(s, o, 32);
  if (lane == 0) bit[row] = s + b_out[0];
}

// ---------------------------------------------------------------------------
// outs.T layout: d_out[row*SEQ + t] = bit[t][row]
// ---------------------------------------------------------------------------
__global__ __launch_bounds__(256)
void gather_kernel(const float* __restrict__ bits,  // SEQ x B
                   float* __restrict__ out)         // B x SEQ
{
  const long total = (long)SEQ * BATCH;
  for (long i = blockIdx.x * (long)blockDim.x + threadIdx.x; i < total;
       i += (long)gridDim.x * blockDim.x) {
    const long t = i / BATCH, row = i % BATCH;
    out[row * SEQ + t] = bits[t * (long)BATCH + row];
  }
}

// ---------------------------------------------------------------------------
extern "C" void kernel_launch(void* const* d_in, const int* in_sizes, int n_in,
                              void* d_out, int out_size, void* d_ws, size_t ws_size,
                              hipStream_t stream) {
  const float* X_f     = (const float*)d_in[0];   // B x 256
  const float* Wproj_f = (const float*)d_in[1];   // 512 x 256
  const float* bproj_f = (const float*)d_in[2];   // 512
  const float* Wih_f   = (const float*)d_in[3];   // 1536 x 257
  const float* bih_f   = (const float*)d_in[4];   // 1536
  const float* Whh_f   = (const float*)d_in[5];   // 1536 x 512
  const float* bhh_f   = (const float*)d_in[6];   // 1536
  const float* Wout_f  = (const float*)d_in[7];   // 1 x 512
  const float* bout_f  = (const float*)d_in[8];   // 1
  float* out = (float*)d_out;

  // Workspace carving (all offsets 16B-aligned).
  char* ws = (char*)d_ws;
  size_t off = 0;
  _Float16* X_h     = (_Float16*)(ws + off); off += (size_t)BATCH * INDIM * 2;
  _Float16* Wproj_h = (_Float16*)(ws + off); off += (size_t)HDIM * INDIM * 2;
  _Float16* Wih_h   = (_Float16*)(ws + off); off += (size_t)GDIM * INDIM * 2;
  _Float16* Whh_h   = (_Float16*)(ws + off); off += (size_t)GDIM * HDIM * 2;
  float*    wbit    = (float*)   (ws + off); off += (size_t)GDIM * 4;
  _Float16* gi_h    = (_Float16*)(ws + off); off += (size_t)BATCH * GDIM * 2;
  _Float16* h_buf0  = (_Float16*)(ws + off); off += (size_t)BATCH * HDIM * 2;
  _Float16* h_buf1  = (_Float16*)(ws + off); off += (size_t)BATCH * HDIM * 2;
  float*    bits    = (float*)   (ws + off); off += (size_t)SEQ * BATCH * 4;
  (void)ws_size; (void)in_sizes; (void)n_in; (void)out_size;

  // 1. fp16 conversions
  prep_kernel<<<4096, 256, 0, stream>>>(Wproj_f, Wih_f, Whh_f, X_f,
                                        Wproj_h, Wih_h, Whh_h, wbit, X_h);

  // 2. h0 + gi_const (fused WMMA GEMM, N = 512 + 1536, 32x32 tiles)
  {
    dim3 grid(BATCH / 128, (HDIM + GDIM) / 32);
    proj_gi_kernel<<<grid, 128, 0, stream>>>(X_h, Wproj_h, Wih_h,
                                             bproj_f, bih_f, h_buf0, gi_h);
  }

  // 3. 8 recurrent steps (stream order = grid-wide sync between steps)
  _Float16* hp = h_buf0;
  _Float16* hn = h_buf1;
  for (int t = 0; t < SEQ; ++t) {
    dim3 grid(BATCH / 128, HDIM / 32);
    const float* bit_prev = (t == 0) ? bits : (bits + (size_t)(t - 1) * BATCH);
    gru_step_kernel<<<grid, 128, 0, stream>>>(hp, hn, gi_h, Whh_h, wbit, bhh_f,
                                              bit_prev, t > 0 ? 1 : 0);
    bit_kernel<<<BATCH / 8, 256, 0, stream>>>(hn, Wout_f, bout_f,
                                              bits + (size_t)t * BATCH);
    _Float16* tmp = hp; hp = hn; hn = tmp;
  }

  // 4. transpose bit matrix into output layout (B x SEQ)
  gather_kernel<<<1024, 256, 0, stream>>>(bits, out);
}
